// ContrastHead_32298154066765
// MI455X (gfx1250) — matrix operand order repaired and checked by reading, not compiled, not run on hardware
//
#include <hip/hip_runtime.h>

typedef _Float16 h2   __attribute__((ext_vector_type(2)));
typedef _Float16 v8h  __attribute__((ext_vector_type(8)));
typedef _Float16 v16h __attribute__((ext_vector_type(16)));
typedef float    v8f  __attribute__((ext_vector_type(8)));

constexpr int   KNB      = 31;    // neighbors per point
constexpr int   CFEAT    = 64;    // feature channels
constexpr int   ROWH     = 72;    // LDS row stride in halfs (144B: bank-conflict-free, 16B aligned)
constexpr int   WAVES    = 8;     // waves per block (256 threads, wave32)
constexpr int   BLOCKS   = 512;
constexpr float TEMP_INV = 10.0f; // 1 / temperature(0.1)
constexpr float EPSF     = 1e-8f;

// ---- CDNA5 async memory->LDS copy (ASYNCcnt path), via inline asm (§15.18 opcode 98) ----
// Per-lane: 16B  memory -> LDS.  vdst = LDS byte offset (low 32 bits of flat shared addr),
// vaddr = 64-bit global address pair.
__device__ __forceinline__ void async_row_b128(const _Float16* g, _Float16* l) {
    unsigned lo = (unsigned)(uintptr_t)l;   // LDS aperture: low 32 bits = LDS byte offset
    asm volatile("global_load_async_to_lds_b128 %0, %1, off" :: "v"(lo), "v"(g) : "memory");
}

__device__ __forceinline__ void wait_async0() {
    asm volatile("s_wait_asynccnt 0x0" ::: "memory");
}

// ---------------- kernel 1: f32 -> f16 feature table (halves gathered bytes) ----------------
__global__ void cvt_f16_kernel(const float* __restrict__ f, _Float16* __restrict__ o, int n) {
    int k = blockIdx.x * blockDim.x + threadIdx.x;
    if (k < n) o[k] = (_Float16)f[k];
}

// ---------------- kernel 2: exact f32 squared row norms of the f16 table ----------------
__global__ void __launch_bounds__(256)
norm_kernel(const _Float16* __restrict__ feat, float* __restrict__ norms, int n) {
    int lane = threadIdx.x & 31, wv = threadIdx.x >> 5;
    int p = blockIdx.x * WAVES + wv;
    if (p >= n) return;
    h2 v = ((const h2*)(feat + (size_t)p * CFEAT))[lane];
    float a = (float)v[0], b = (float)v[1];
    float s = a * a + b * b;
#pragma unroll
    for (int off = 16; off > 0; off >>= 1) s += __shfl_xor(s, off, 32);
    if (lane == 0) norms[p] = s;
}

// ---------------- kernel 3: per-point loss; async gather + WMMA dot products ----------------
__global__ void __launch_bounds__(256)
ContrastHead_32298154066765_kernel(const _Float16* __restrict__ feat,
                                   const float* __restrict__ norms,
                                   const int* __restrict__ labels,
                                   const int* __restrict__ nidx,
                                   float* __restrict__ partials, int n) {
    __shared__ __align__(16) _Float16 tile[WAVES][32 * ROWH]; // raw neighbor rows, 32x64 f16
    __shared__ float sbuf[WAVES][32];
    __shared__ float wsum[WAVES][2];

    const int lane = threadIdx.x & 31;
    const int wv   = threadIdx.x >> 5;
    const int l15  = lane & 15;
    const int hi16 = lane >> 4;
    const int sub  = lane & 7;                    // 16B chunk within a 128B row
    const int waveGlobal = blockIdx.x * WAVES + wv;
    const int waveStride = gridDim.x * WAVES;

    float accL = 0.0f, accC = 0.0f;

    for (int i = waveGlobal; i < n; i += waveStride) {
        // ---- one coalesced lane-parallel neighbor-index load (replaces 31 serial loads) ----
        const int* nrow = nidx + (size_t)i * KNB;
        int nbl = nrow[lane < KNB ? lane : KNB - 1];

        // ---- async gather: 32 rows x 128B, 4 rows per b128 instruction, no VGPR round-trip
        #pragma unroll
        for (int g = 0; g < 8; ++g) {
            int j   = g * 4 + (lane >> 3);
            int nb  = __shfl(nbl, j < KNB ? j : KNB - 1, 32);
            async_row_b128(feat + (size_t)nb * CFEAT + sub * 8,
                           &tile[wv][j * ROWH + sub * 8]);
        }

        // ---- overlap small gathers with the async transfer ----
        h2  q2   = ((const h2*)(feat + (size_t)i * CFEAT))[lane];
        int myl  = labels[i];
        float qn = norms[i];
        int  nlab = labels[nbl];
        float nn  = norms[nbl];

        wait_async0();  // all 32 rows resident in LDS

        // ---- 4 WMMAs: dot(n_m, q). A = neighbor tile (documented 16-bit A 16x32 layout),
        //      B = q broadcast across all 16 columns (element s of half h holds k in table order, as A) ----
        const _Float16* base = &tile[wv][0];
        v8f acc0 = {0.f, 0.f, 0.f, 0.f, 0.f, 0.f, 0.f, 0.f};
        v8f acc1 = {0.f, 0.f, 0.f, 0.f, 0.f, 0.f, 0.f, 0.f};
        union { h2 h; unsigned u; } qcv; qcv.h = q2;
#pragma unroll
        for (int kk = 0; kk < 2; ++kk) {
            // B: element s of lane half h holds q[kk*32 + k], k in table order (same as A)
            union { v16h v; unsigned u[8]; } bu;
#pragma unroll
            for (int t = 0; t < 8; ++t)
                bu.u[t] = (unsigned)__shfl((int)qcv.u, kk * 16 + hi16 * 4 + (t < 4 ? t : t + 4), 32);

            const _Float16* r0 = base + (size_t)l15 * ROWH + kk * 32 + hi16 * 8;
            v8h a0lo = *(const v8h*)(r0);
            v8h a0hi = *(const v8h*)(r0 + 16);
            v16h A0 = __builtin_shufflevector(a0lo, a0hi, 0,1,2,3,4,5,6,7,8,9,10,11,12,13,14,15);
            acc0 = __builtin_amdgcn_wmma_f32_16x16x32_f16(false, A0, false, bu.v,
                                                          (short)0, acc0, false, false);
            const _Float16* r1 = base + (size_t)(16 + l15) * ROWH + kk * 32 + hi16 * 8;
            v8h a1lo = *(const v8h*)(r1);
            v8h a1hi = *(const v8h*)(r1 + 16);
            v16h A1 = __builtin_shufflevector(a1lo, a1hi, 0,1,2,3,4,5,6,7,8,9,10,11,12,13,14,15);
            acc1 = __builtin_amdgcn_wmma_f32_16x16x32_f16(false, A1, false, bu.v,
                                                          (short)0, acc1, false, false);
        }

        // C/D layout: VGPR r holds out[M = r + 8*hi16][N = l15]; columns identical
        if (l15 == 0) {
#pragma unroll
            for (int r = 0; r < 8; ++r) {
                sbuf[wv][hi16 * 8 + r]      = acc0[r];
                sbuf[wv][16 + hi16 * 8 + r] = acc1[r];
            }
        }
        __builtin_amdgcn_wave_barrier();
        float dot = sbuf[wv][lane];

        // ---- masked softmax over 31 neighbors (wave32 shfl reductions) ----
        bool  valid = lane < KNB;
        bool  pm    = valid && (nlab == myl);
        float d2    = fmaxf(qn + nn - 2.0f * dot, 0.0f);
        float dist  = sqrtf(d2) + EPSF;
        float logit = valid ? -dist : -3.0e38f;

        float m = logit;
#pragma unroll
        for (int off = 16; off > 0; off >>= 1) m = fmaxf(m, __shfl_xor(m, off, 32));

        float e   = valid ? expf((logit - m) * TEMP_INV) : 0.0f;
        float pos = pm ? e : 0.0f;
        float neg = e;
#pragma unroll
        for (int off = 16; off > 0; off >>= 1) {
            pos += __shfl_xor(pos, off, 32);
            neg += __shfl_xor(neg, off, 32);
        }
        int cnt = __popcll(__ballot(pm));
        if (lane == 0 && cnt > 0 && cnt < KNB) {
            accL += -logf(pos / neg + EPSF);
            accC += 1.0f;
        }
    }

    if (lane == 0) { wsum[wv][0] = accL; wsum[wv][1] = accC; }
    __syncthreads();
    if (threadIdx.x == 0) {
        float L = 0.f, Cn = 0.f;
        for (int w = 0; w < WAVES; ++w) { L += wsum[w][0]; Cn += wsum[w][1]; }
        partials[2 * blockIdx.x]     = L;
        partials[2 * blockIdx.x + 1] = Cn;
    }
}

// ---------------- kernel 4: deterministic final reduction ----------------
__global__ void reduce_kernel(const float* __restrict__ partials, float* __restrict__ out, int nb) {
    __shared__ float sl[256], sc[256];
    int t = threadIdx.x;
    float L = 0.f, Cc = 0.f;
    for (int k = t; k < nb; k += 256) { L += partials[2 * k]; Cc += partials[2 * k + 1]; }
    sl[t] = L; sc[t] = Cc;
    __syncthreads();
    for (int off = 128; off > 0; off >>= 1) {
        if (t < off) { sl[t] += sl[t + off]; sc[t] += sc[t + off]; }
        __syncthreads();
    }
    if (t == 0) out[0] = sl[0] / fmaxf(sc[0], 1.0f);   // WEIGHT == 1.0
}

extern "C" void kernel_launch(void* const* d_in, const int* in_sizes, int n_in,
                              void* d_out, int out_size, void* d_ws, size_t ws_size,
                              hipStream_t stream) {
    const float* features = (const float*)d_in[0];
    const int*   labels   = (const int*)d_in[1];
    const int*   nidx     = (const int*)d_in[2];
    const int NC = in_sizes[0];        // N * 64
    const int N  = in_sizes[1];

    _Float16* feat16 = (_Float16*)d_ws;                                   // 12.8 MB, L2-resident
    float* norms     = (float*)((char*)d_ws + (size_t)NC * sizeof(_Float16));
    float* partials  = norms + N;

    cvt_f16_kernel<<<(NC + 255) / 256, 256, 0, stream>>>(features, feat16, NC);
    norm_kernel<<<(N + WAVES - 1) / WAVES, 256, 0, stream>>>(feat16, norms, N);
    ContrastHead_32298154066765_kernel<<<BLOCKS, 256, 0, stream>>>(feat16, norms, labels, nidx,
                                                                   partials, N);
    reduce_kernel<<<1, 256, 0, stream>>>(partials, (float*)d_out, BLOCKS);
}
